// HMM_7370163880597
// MI455X (gfx1250) — compile-verified, matrix operand order checked by reading, not run on hardware
//
#include <hip/hip_runtime.h>
#include <hip/hip_bf16.h>
#include <cstdint>
#include <cstddef>

// Problem dims (fixed by the reference)
#define HMM_B 32
#define HMM_T 256
#define HMM_H 1024
#define HMM_V 50257
#define HMM_NBLK 16            // persistent grid: 16 blocks x 4 waves = 64 waves

typedef __attribute__((ext_vector_type(16))) _Float16 v16h;
typedef __attribute__((ext_vector_type(8)))  _Float16 v8h;
typedef __attribute__((ext_vector_type(8)))  float    v8f;

union Frag16 { v16h v; v8h h[2]; };

// ---------------------------------------------------------------------------
// A16[i*H + j] = exp(alpha[i,j])   (log-softmax rows -> values in [0,1], f16 ok)
// ---------------------------------------------------------------------------
__global__ void hmm_exp_alpha(const float* __restrict__ alpha,
                              _Float16* __restrict__ A16) {
    int idx = blockIdx.x * blockDim.x + threadIdx.x;   // grid covers H*H exactly
    A16[idx] = (_Float16)__expf(alpha[idx]);
}

// ---------------------------------------------------------------------------
// y0[b,h] = beta[h, x[b, T-1]]
// ---------------------------------------------------------------------------
__global__ void hmm_init_y(const float* __restrict__ beta,
                           const int* __restrict__ x,
                           float* __restrict__ y) {
    int idx = blockIdx.x * blockDim.x + threadIdx.x;   // over B*H
    int b = idx >> 10;
    int h = idx & (HMM_H - 1);
    int xv = x[b * HMM_T + (HMM_T - 1)];
    y[idx] = beta[(size_t)h * HMM_V + xv];
}

// ---------------------------------------------------------------------------
// One-time prep for t=T-2: m[b] = max_h y[b,h];  P16[b,h] = exp(y[b,h]-m[b])
// ---------------------------------------------------------------------------
__global__ void hmm_prep(const float* __restrict__ y,
                         float* __restrict__ m,
                         _Float16* __restrict__ P) {
    __shared__ float red[256];
    const int b   = blockIdx.x;
    const int tid = threadIdx.x;
    const float* yr = y + b * HMM_H;

    float mx = -3.4028235e38f;
    for (int h = tid; h < HMM_H; h += 256) mx = fmaxf(mx, yr[h]);
    red[tid] = mx;
    __syncthreads();
    for (int s = 128; s > 0; s >>= 1) {
        if (tid < s) red[tid] = fmaxf(red[tid], red[tid + s]);
        __syncthreads();
    }
    float mb = red[0];
    if (tid == 0) m[b] = mb;
    for (int h = tid; h < HMM_H; h += 256)
        P[b * HMM_H + h] = (_Float16)__expf(yr[h] - mb);
}

// ---------------------------------------------------------------------------
// Zero the grid-barrier counters (must run every launch; ws is not re-poisoned)
// ---------------------------------------------------------------------------
__global__ void hmm_zero_sync(unsigned* c) { c[threadIdx.x] = 0u; }

// ---------------------------------------------------------------------------
// Sense-reversing grid barrier over HMM_NBLK blocks (device-scope atomics).
// ---------------------------------------------------------------------------
__device__ __forceinline__ void grid_barrier(unsigned* cnt, unsigned* gen,
                                             unsigned nblk) {
    __syncthreads();
    if (threadIdx.x == 0) {
        __threadfence();   // make this block's stores visible device-wide
        unsigned g = __hip_atomic_load(gen, __ATOMIC_RELAXED,
                                       __HIP_MEMORY_SCOPE_AGENT);
        unsigned a = __hip_atomic_fetch_add(cnt, 1u, __ATOMIC_ACQ_REL,
                                            __HIP_MEMORY_SCOPE_AGENT);
        if (a == nblk - 1u) {
            __hip_atomic_store(cnt, 0u, __ATOMIC_RELAXED,
                               __HIP_MEMORY_SCOPE_AGENT);
            __hip_atomic_fetch_add(gen, 1u, __ATOMIC_RELEASE,
                                   __HIP_MEMORY_SCOPE_AGENT);
        } else {
            while (__hip_atomic_load(gen, __ATOMIC_ACQUIRE,
                                     __HIP_MEMORY_SCOPE_AGENT) == g) {
                __builtin_amdgcn_s_sleep(1);
            }
        }
        __threadfence();   // invalidate stale cached lines before re-reading
    }
    __syncthreads();
}

// ---------------------------------------------------------------------------
// Persistent recursion kernel: runs all 255 steps with grid barriers.
// Per step (wave = one 16-wide N-tile, both 16-row M-tiles):
//   prefetch e_t[b,i] = beta[i, x[b,t]]  (HBM gather, hidden behind GEMM)
//   Z = P · A16^T via v_wmma_f32_16x16x32_f16 (K=1024 in chunks of 32)
//   y = log(Z) + m_old + e_t                  (registers only)
//   cross-lane max -> block partials -> grid sync -> new m (LDS)
//   P_new = exp(y - m_new) written from registers -> grid sync
// ---------------------------------------------------------------------------
__global__ __launch_bounds__(128) void hmm_persist(
    const _Float16* __restrict__ A16,   // H x H  exp(alpha[i,j]) at [i*H+j]
    _Float16*       __restrict__ P,     // B x H  exp(y - m), ping buffer
    const float*    __restrict__ m_init,// B      initial per-batch max
    const float*    __restrict__ beta,  // H x V
    const int*      __restrict__ x,     // B x T
    float*          __restrict__ yfinal,// B x H  y at t==0 (for final LSE)
    float*          __restrict__ gpartial, // NBLK x B partial maxima
    unsigned*       __restrict__ syncv) // [0]=cnt, [1]=gen
{
    __shared__ float mlds[HMM_B];
    __shared__ int   xlds[HMM_B];
    __shared__ float pmax[4][HMM_B];

    const int tid   = threadIdx.x;
    const int lane  = tid & 31;
    const int wave  = tid >> 5;
    const int ntile = blockIdx.x * 4 + wave;          // 0..63
    const int nlo   = lane & 15;
    const int khalf = (lane >> 4) << 3;               // 0 or 8 (K half / M half)
    const int icol  = (ntile << 4) + nlo;             // output column i

    unsigned* cnt = syncv + 0;
    unsigned* gen = syncv + 1;

    if (tid < HMM_B) mlds[tid] = m_init[tid];

    const _Float16* prow0 = P + (size_t)nlo        * HMM_H;  // b = 0..15
    const _Float16* prow1 = P + (size_t)(16 + nlo) * HMM_H;  // b = 16..31
    const _Float16* bcol  = A16 + (size_t)icol * HMM_H;      // contiguous in K
    const float*    bgath = beta + (size_t)icol * HMM_V;     // gather base

    for (int t = HMM_T - 2; t >= 0; --t) {
        if (tid < HMM_B) xlds[tid] = x[tid * HMM_T + t];
        __syncthreads();               // mlds/xlds ready; P ready (prev barrier)

        // Issue the scattered beta gathers NOW so their HBM latency overlaps
        // the entire K-loop below (they are independent of the GEMM).
        float e0[8], e1[8];
        #pragma unroll
        for (int r = 0; r < 8; ++r) {
            e0[r] = bgath[xlds[r + khalf]];
            e1[r] = bgath[xlds[16 + r + khalf]];
        }

        v8f c0 = {};
        v8f c1 = {};
        #pragma unroll 4
        for (int kk = 0; kk < HMM_H; kk += 32) {
            const int kb = kk + khalf;
            Frag16 a0, a1, bb;
            a0.h[0] = *(const v8h*)(prow0 + kb);
            a0.h[1] = *(const v8h*)(prow0 + kb + 16);
            a1.h[0] = *(const v8h*)(prow1 + kb);
            a1.h[1] = *(const v8h*)(prow1 + kb + 16);
            bb.h[0] = *(const v8h*)(bcol  + kb);
            bb.h[1] = *(const v8h*)(bcol  + kb + 16);
            c0 = __builtin_amdgcn_wmma_f32_16x16x32_f16(
                     false, a0.v, false, bb.v, (short)0, c0, false, false);
            c1 = __builtin_amdgcn_wmma_f32_16x16x32_f16(
                     false, a1.v, false, bb.v, (short)0, c1, false, false);
        }

        // Epilogue in registers: y = log(Z) + m_old + e_t
        // D layout: lane -> col N = lane&15; VGPR r -> row M = r + khalf
        float yv0[8], yv1[8];
        #pragma unroll
        for (int r = 0; r < 8; ++r) {
            const int b0 = r + khalf;
            const int b1 = 16 + r + khalf;
            yv0[r] = __logf(c0[r]) + mlds[b0] + e0[r];
            yv1[r] = __logf(c1[r]) + mlds[b1] + e1[r];
        }
        if (t == 0) {
            #pragma unroll
            for (int r = 0; r < 8; ++r) {
                yfinal[(r + khalf)      * HMM_H + icol] = yv0[r];
                yfinal[(16 + r + khalf) * HMM_H + icol] = yv1[r];
            }
        }

        // Cross-lane max over the 16 columns this wave owns (xor masks 1..8
        // stay within each 16-lane half, which is exactly the b-row group).
        float pm0[8], pm1[8];
        #pragma unroll
        for (int r = 0; r < 8; ++r) { pm0[r] = yv0[r]; pm1[r] = yv1[r]; }
        #pragma unroll
        for (int mask = 1; mask <= 8; mask <<= 1) {
            #pragma unroll
            for (int r = 0; r < 8; ++r) {
                pm0[r] = fmaxf(pm0[r], __shfl_xor(pm0[r], mask, 32));
                pm1[r] = fmaxf(pm1[r], __shfl_xor(pm1[r], mask, 32));
            }
        }
        if (nlo == 0) {
            #pragma unroll
            for (int r = 0; r < 8; ++r) {
                pmax[wave][r + khalf]      = pm0[r];
                pmax[wave][16 + r + khalf] = pm1[r];
            }
        }
        __syncthreads();
        if (tid < HMM_B) {
            float bm = fmaxf(fmaxf(pmax[0][tid], pmax[1][tid]),
                             fmaxf(pmax[2][tid], pmax[3][tid]));
            gpartial[blockIdx.x * HMM_B + tid] = bm;
        }
        grid_barrier(cnt, gen, HMM_NBLK);

        // New per-batch max from all block partials (computed redundantly)
        if (tid < HMM_B) {
            float mm = -3.4028235e38f;
            #pragma unroll
            for (int blk = 0; blk < HMM_NBLK; ++blk)
                mm = fmaxf(mm, gpartial[blk * HMM_B + tid]);
            mlds[tid] = mm;
        }
        __syncthreads();

        // New P written straight from registers
        #pragma unroll
        for (int r = 0; r < 8; ++r) {
            const int b0 = r + khalf;
            const int b1 = 16 + r + khalf;
            P[b0 * HMM_H + icol] = (_Float16)__expf(yv0[r] - mlds[b0]);
            P[b1 * HMM_H + icol] = (_Float16)__expf(yv1[r] - mlds[b1]);
        }
        grid_barrier(cnt, gen, HMM_NBLK);
    }
}

// ---------------------------------------------------------------------------
// out[b] = logsumexp_h(gamma[h] + y[b,h])
// ---------------------------------------------------------------------------
__global__ void hmm_final(const float* __restrict__ y,
                          const float* __restrict__ gamma,
                          float* __restrict__ out) {
    __shared__ float red[256];
    const int b   = blockIdx.x;
    const int tid = threadIdx.x;

    float mx = -3.4028235e38f;
    for (int h = tid; h < HMM_H; h += 256)
        mx = fmaxf(mx, gamma[h] + y[b * HMM_H + h]);
    red[tid] = mx;
    __syncthreads();
    for (int s = 128; s > 0; s >>= 1) {
        if (tid < s) red[tid] = fmaxf(red[tid], red[tid + s]);
        __syncthreads();
    }
    const float mb = red[0];
    __syncthreads();

    float sm = 0.0f;
    for (int h = tid; h < HMM_H; h += 256)
        sm += __expf(gamma[h] + y[b * HMM_H + h] - mb);
    red[tid] = sm;
    __syncthreads();
    for (int s = 128; s > 0; s >>= 1) {
        if (tid < s) red[tid] += red[tid + s];
        __syncthreads();
    }
    if (tid == 0) out[b] = __logf(red[0]) + mb;
}

// ---------------------------------------------------------------------------
// Workspace layout (16B-aligned offsets), ~2.33 MB total:
//   A16      : H*H f16         (2 MB)
//   y0       : B*H f32         (128 KB)  initial y = e[:, T-1]
//   yfin     : B*H f32         (128 KB)  y at t==0
//   P        : B*H f16         (64 KB)
//   m        : B   f32
//   gpartial : NBLK*B f32      (2 KB)
//   syncv    : 2 u32
// ---------------------------------------------------------------------------
extern "C" void kernel_launch(void* const* d_in, const int* in_sizes, int n_in,
                              void* d_out, int out_size, void* d_ws, size_t ws_size,
                              hipStream_t stream) {
    const int*   x     = (const int*)d_in[0];
    const float* alpha = (const float*)d_in[1];
    const float* beta  = (const float*)d_in[2];
    const float* gamma = (const float*)d_in[3];
    float* out = (float*)d_out;

    char* ws = (char*)d_ws;
    _Float16* A16   = (_Float16*)ws; ws += (size_t)HMM_H * HMM_H * sizeof(_Float16);
    float*    y0    = (float*)ws;    ws += (size_t)HMM_B * HMM_H * sizeof(float);
    float*    yfin  = (float*)ws;    ws += (size_t)HMM_B * HMM_H * sizeof(float);
    _Float16* P     = (_Float16*)ws; ws += (size_t)HMM_B * HMM_H * sizeof(_Float16);
    float*    m     = (float*)ws;    ws += 64;
    float*    gpart = (float*)ws;    ws += (size_t)HMM_NBLK * HMM_B * sizeof(float);
    unsigned* syncv = (unsigned*)ws;

    hmm_exp_alpha<<<(HMM_H * HMM_H) / 256, 256, 0, stream>>>(alpha, A16);
    hmm_init_y<<<(HMM_B * HMM_H) / 256, 256, 0, stream>>>(beta, x, y0);
    hmm_prep<<<HMM_B, 256, 0, stream>>>(y0, m, P);
    hmm_zero_sync<<<1, 2, 0, stream>>>(syncv);
    hmm_persist<<<HMM_NBLK, 128, 0, stream>>>(A16, P, m, beta, x, yfin,
                                              gpart, syncv);
    hmm_final<<<HMM_B, 256, 0, stream>>>(yfin, gamma, out);
}
